// MachineNodeEmbeddingLayer_81862076662463
// MI455X (gfx1250) — matrix operand order repaired and checked by reading, not compiled, not run on hardware
//
#include <hip/hip_runtime.h>
#include <hip/hip_bf16.h>

typedef __attribute__((ext_vector_type(16))) _Float16 v16h;
typedef __attribute__((ext_vector_type(8)))  _Float16 v8h;
typedef __attribute__((ext_vector_type(8)))  float    v8f;

#define HEADS 4
#define DIMD  32
#define KDIM  128   // input feature dim = H*D = 128 as well

// ---------------------------------------------------------------------------
// GEMM: Out[M,128] = X[M,128] @ W[128,128], f16 WMMA with f32 accumulate.
// Block: 256 threads (8 waves), 128 rows per block. Wave w -> rows [16w,16w+16).
// ---------------------------------------------------------------------------
__launch_bounds__(256)
__global__ void gemm128_wmma_kernel(const float* __restrict__ X,
                                    const float* __restrict__ W,
                                    float* __restrict__ Out, int M) {
    __shared__ __align__(32) _Float16 sA[128 * 128];  // X tile rows, row-major f16
    __shared__ __align__(32) _Float16 sB[128 * 128];  // W packed as WMMA B fragments

    const int t = threadIdx.x;
    const int blockRow = blockIdx.x * 128;

    // Stage A: 128 rows x 128 cols f32 -> f16 (clamp OOB rows; their stores are guarded)
    for (int i = t; i < 128 * 128; i += 256) {
        int r = i >> 7, c = i & 127;
        int gr = blockRow + r;
        if (gr >= M) gr = M - 1;
        sA[i] = (_Float16)X[(size_t)gr * 128 + c];
    }
    // Stage B: pack W into per-fragment layout.
    // Fragment (ks, nt): 32 lanes x 16 halves. Lane group g = lane>>4 holds
// K = ks*32 + (i<8 ? g*8 + i : 16 + g*8 + (i-8)) (i = half index), column n = nt*16 + (lane&15).
    for (int p = t; p < 128 * 128; p += 256) {
        int i    = p & 15;
        int lane = (p >> 4) & 31;
        int nt   = (p >> 9) & 7;
        int ks   = (p >> 12) & 3;
        int g = lane >> 4, nl = lane & 15;
        int k = ks * 32 + (i < 8 ? g * 8 + i : 16 + g * 8 + (i - 8));
        int n = nt * 16 + nl;
        sB[p] = (_Float16)W[k * 128 + n];
    }
    __syncthreads();

    const int wave = t >> 5;
    const int lane = t & 31;
    const int g = lane >> 4;       // lane group
    const int r = lane & 15;
    const int rowl = wave * 16 + r;

    v8f acc[8] = {};

    for (int ks = 0; ks < 4; ks++) {
        // A fragment (16x32 f16): halves 0..7 = K[ks*32 + g*8 .. +8),
        //                         halves 8..15 = K[ks*32 + 16 + g*8 .. +8)
        const _Float16* pa = &sA[rowl * 128 + ks * 32 + g * 8];
        v8h lo = *(const v8h*)pa;
        v8h hi = *(const v8h*)(pa + 16);
        v16h a = __builtin_shufflevector(lo, hi, 0, 1, 2, 3, 4, 5, 6, 7,
                                                 8, 9, 10, 11, 12, 13, 14, 15);
        for (int n = 0; n < 8; n++) {
            v16h b = *(const v16h*)&sB[(((ks * 8 + n) * 32) + lane) * 16];
            acc[n] = __builtin_amdgcn_wmma_f32_16x16x32_f16(
                false, a, false, b, (short)0, acc[n], false, false);
        }
    }

    // C/D layout: lane group g, VGPR v -> row M = g*8 + v, col N = lane&15
    for (int n = 0; n < 8; n++) {
        for (int v = 0; v < 8; v++) {
            int grow = blockRow + wave * 16 + g * 8 + v;
            if (grow < M) Out[(size_t)grow * 128 + n * 16 + r] = acc[n][v];
        }
    }
}

// ---------------------------------------------------------------------------
// Per-edge GATv2 scores: one wave (32 lanes) per edge; lane = d index.
// ex = exp(score) stored per (edge, head); denom accumulated via f32 atomics.
// Masked edges contribute exactly 0 (== exp(-1e9) in the reference).
// ---------------------------------------------------------------------------
__launch_bounds__(256)
__global__ void edge_score_kernel(const float* __restrict__ m_src,
                                  const float* __restrict__ m_dst,
                                  const int* __restrict__ esrc,
                                  const int* __restrict__ edst,
                                  const float* __restrict__ eattr,
                                  const unsigned char* __restrict__ mask,
                                  const float* __restrict__ W_edge,
                                  const float* __restrict__ att,
                                  float* __restrict__ ex_om,
                                  float* __restrict__ denom, int E) {
    const int lane = threadIdx.x & 31;
    const int w  = blockIdx.x * (blockDim.x >> 5) + (threadIdx.x >> 5);
    const int nw = gridDim.x * (blockDim.x >> 5);

    float we0[HEADS], we1[HEADS], attv[HEADS];
    for (int h = 0; h < HEADS; h++) {
        we0[h]  = W_edge[h * DIMD + lane];
        we1[h]  = W_edge[128 + h * DIMD + lane];
        attv[h] = att[h * DIMD + lane];
    }

    for (int e = w; e < E; e += nw) {
        int src = esrc[e], dst = edst[e];
        float a0 = eattr[2 * e], a1 = eattr[2 * e + 1];
        bool mk = mask[e] != 0;
        size_t sb = (size_t)src * 128, db = (size_t)dst * 128;
        float myex = 0.f;
        for (int h = 0; h < HEADS; h++) {
            float gv = m_src[sb + h * DIMD + lane] + m_dst[db + h * DIMD + lane]
                     + a0 * we0[h] + a1 * we1[h];
            gv = gv > 0.f ? gv : 0.2f * gv;          // leaky_relu, slope 0.2
            float p = gv * attv[h];
            p += __shfl_xor(p, 16, 32);
            p += __shfl_xor(p, 8, 32);
            p += __shfl_xor(p, 4, 32);
            p += __shfl_xor(p, 2, 32);
            p += __shfl_xor(p, 1, 32);
            float ex = mk ? __expf(p) : 0.f;
            if (lane == h) myex = ex;
        }
        if (lane < HEADS) {
            ex_om[(size_t)e * HEADS + lane] = myex;
            if (myex > 0.f) atomicAdd(&denom[(size_t)dst * HEADS + lane], myex);
        }
    }
}

// Self-loop scores (no edge features, no mask): ex_sl[n][h] = exp(score)
__launch_bounds__(256)
__global__ void self_score_kernel(const float* __restrict__ m_self,
                                  const float* __restrict__ m_dst,
                                  const float* __restrict__ att,
                                  float* __restrict__ ex_sl, int NM) {
    const int lane = threadIdx.x & 31;
    const int w  = blockIdx.x * (blockDim.x >> 5) + (threadIdx.x >> 5);
    const int nw = gridDim.x * (blockDim.x >> 5);
    float attv[HEADS];
    for (int h = 0; h < HEADS; h++) attv[h] = att[h * DIMD + lane];

    for (int n = w; n < NM; n += nw) {
        size_t nb = (size_t)n * 128;
        float myex = 0.f;
        for (int h = 0; h < HEADS; h++) {
            float gv = m_self[nb + h * DIMD + lane] + m_dst[nb + h * DIMD + lane];
            gv = gv > 0.f ? gv : 0.2f * gv;
            float p = gv * attv[h];
            p += __shfl_xor(p, 16, 32);
            p += __shfl_xor(p, 8, 32);
            p += __shfl_xor(p, 4, 32);
            p += __shfl_xor(p, 2, 32);
            p += __shfl_xor(p, 1, 32);
            float ex = __expf(p);
            if (lane == h) myex = ex;
        }
        if (lane < HEADS) ex_sl[(size_t)n * HEADS + lane] = myex;
    }
}

// Aggregate: acc[dst][h][d] += m_src[src][h][d] * ex_om[e][h]  (f32 atomics, L2-resident)
__launch_bounds__(256)
__global__ void aggregate_kernel(const float* __restrict__ m_src,
                                 const int* __restrict__ esrc,
                                 const int* __restrict__ edst,
                                 const float* __restrict__ ex_om,
                                 float* __restrict__ acc, int E) {
    const int lane = threadIdx.x & 31;
    const int w  = blockIdx.x * (blockDim.x >> 5) + (threadIdx.x >> 5);
    const int nw = gridDim.x * (blockDim.x >> 5);

    for (int e = w; e < E; e += nw) {
        int src = esrc[e], dst = edst[e];
        float exv = (lane < HEADS) ? ex_om[(size_t)e * HEADS + lane] : 0.f;
        size_t sb = (size_t)src * 128, db = (size_t)dst * 128;
        for (int h = 0; h < HEADS; h++) {
            float exh = __shfl(exv, h, 32);
            if (exh != 0.f) {   // uniform across wave; skip masked edges
                float v = m_src[sb + h * DIMD + lane] * exh;
                atomicAdd(&acc[db + h * DIMD + lane], v);
            }
        }
    }
}

// Finalize: add self-loop numerator, divide by denom(+self), head mean, bias, sigmoid
__launch_bounds__(256)
__global__ void finalize_kernel(const float* __restrict__ acc,
                                const float* __restrict__ m_self,
                                const float* __restrict__ ex_sl,
                                const float* __restrict__ denom,
                                const float* __restrict__ bias,
                                float* __restrict__ out, int NM) {
    int t = blockIdx.x * blockDim.x + threadIdx.x;
    if (t >= NM * DIMD) return;
    int n = t >> 5, d = t & 31;
    size_t nb = (size_t)n * 128;
    float s = 0.f;
    for (int h = 0; h < HEADS; h++) {
        float exs = ex_sl[(size_t)n * HEADS + h];
        float num = acc[nb + h * DIMD + d] + m_self[nb + h * DIMD + d] * exs;
        float den = denom[(size_t)n * HEADS + h] + exs;
        s += num / den;
    }
    float v = 0.25f * s + bias[d];        // mean over 4 heads + bias
    out[t] = 1.f / (1.f + __expf(-v));    // sigmoid
}

extern "C" void kernel_launch(void* const* d_in, const int* in_sizes, int n_in,
                              void* d_out, int out_size, void* d_ws, size_t ws_size,
                              hipStream_t stream) {
    const float* x_op   = (const float*)d_in[0];
    const float* x_m    = (const float*)d_in[1];
    const float* W_src  = (const float*)d_in[2];
    const float* W_self = (const float*)d_in[3];
    const float* W_dst  = (const float*)d_in[4];
    const float* W_edge = (const float*)d_in[5];
    const float* att    = (const float*)d_in[6];
    const float* bias   = (const float*)d_in[7];
    const int* esrc     = (const int*)d_in[8];
    const int* edst     = (const int*)d_in[9];
    const float* eattr  = (const float*)d_in[10];
    const unsigned char* mask = (const unsigned char*)d_in[11];

    const int N_OP = in_sizes[0] / 128;
    const int NM   = in_sizes[1] / 128;
    const int E    = in_sizes[8];

    char* ws = (char*)d_ws;
    float* m_src  = (float*)ws; ws += (size_t)N_OP * 128 * sizeof(float);
    float* m_self = (float*)ws; ws += (size_t)NM * 128 * sizeof(float);
    float* m_dst  = (float*)ws; ws += (size_t)NM * 128 * sizeof(float);
    float* ex_om  = (float*)ws; ws += (size_t)E * HEADS * sizeof(float);
    float* ex_sl  = (float*)ws; ws += (size_t)NM * HEADS * sizeof(float);
    float* denom  = (float*)ws; ws += (size_t)NM * HEADS * sizeof(float);
    float* acc    = (float*)ws; ws += (size_t)NM * 128 * sizeof(float);

    hipMemsetAsync(denom, 0, (size_t)NM * HEADS * sizeof(float), stream);
    hipMemsetAsync(acc,   0, (size_t)NM * 128 * sizeof(float), stream);

    // Projections (WMMA f16 -> f32)
    gemm128_wmma_kernel<<<(N_OP + 127) / 128, 256, 0, stream>>>(x_op, W_src, m_src, N_OP);
    gemm128_wmma_kernel<<<(NM + 127) / 128, 256, 0, stream>>>(x_m, W_self, m_self, NM);
    gemm128_wmma_kernel<<<(NM + 127) / 128, 256, 0, stream>>>(x_m, W_dst, m_dst, NM);

    // Edge + self-loop scores
    edge_score_kernel<<<4096, 256, 0, stream>>>(m_src, m_dst, esrc, edst, eattr, mask,
                                                W_edge, att, ex_om, denom, E);
    self_score_kernel<<<(NM + 7) / 8, 256, 0, stream>>>(m_self, m_dst, att, ex_sl, NM);

    // Weighted aggregation via L2 atomics
    aggregate_kernel<<<4096, 256, 0, stream>>>(m_src, esrc, edst, ex_om, acc, E);

    // Normalize, head mean, bias, sigmoid
    finalize_kernel<<<(NM * DIMD + 255) / 256, 256, 0, stream>>>(acc, m_self, ex_sl,
                                                                 denom, bias,
                                                                 (float*)d_out, NM);
}
